// EncoderRNN_68264210203009
// MI455X (gfx1250) — compile-verified, matrix operand order checked by reading, not run on hardware
//
#include <hip/hip_runtime.h>
#include <hip/hip_bf16.h>
#include <stdint.h>

#define T_DIM 512
#define B_DIM 128
#define E_DIM 512
#define H_DIM 512
#define MEM_DIM 500
#define MEMP 512      // MEM padded to K multiple of 32
#define CTX_DIM 300
#define CTXP 320      // CTX padded to K multiple of 32 (and N multiple of 16)
#define G4H 2048      // 4*H
#define AK (E_DIM + H_DIM + CTXP)   // 1344: staged activation row length

typedef __attribute__((ext_vector_type(16))) __bf16 v16bf;
typedef __attribute__((ext_vector_type(8)))  __bf16 bf16x8;
typedef __attribute__((ext_vector_type(2)))  __bf16 bf16x2;
typedef __attribute__((ext_vector_type(8)))  float  v8f;

// ---------------------------------------------------------------------------
// WMMA fragment loader (wave32, 16x16x32 bf16, CDNA5 ISA 7.12.2):
// lane l<16 holds row l with K in [0..7]∪[16..23]; lane l+16 holds row l with
// K in [8..15]∪[24..31]  ->  two contiguous 16-byte runs per lane.
// A reads activations row-major [M][K]; B reads weights row-major [N][K]
// (B[k][n] == W[n][k]), so one loader serves both operands (global or LDS).
// ---------------------------------------------------------------------------
__device__ __forceinline__ v16bf load_frag_bf16(const __bf16* base, int ld) {
  const int lane = threadIdx.x & 31;
  const int mn = lane & 15, half = lane >> 4;
  const __bf16* row = base + (size_t)mn * ld + 8 * half;
  const bf16x8 lo = *reinterpret_cast<const bf16x8*>(row);        // K run 0
  const bf16x8 hi = *reinterpret_cast<const bf16x8*>(row + 16);   // K run 1
  v16bf f;
#pragma unroll
  for (int i = 0; i < 8; ++i) { f[i] = lo[i]; f[i + 8] = hi[i]; }
  return f;
}

__device__ __forceinline__ v8f wmma_bf16(v16bf a, v16bf b, v8f c) {
  return __builtin_amdgcn_wmma_f32_16x16x32_bf16(false, a, false, b,
                                                 (short)0, c, false, false);
}

__device__ __forceinline__ float sigm(float x) { return 1.f / (1.f + __expf(-x)); }

// ---------------------------------------------------------------------------
// One-time weight repack: fp32 -> bf16 row-major [N][K], K padded, pads zero.
// ---------------------------------------------------------------------------
__global__ void pack_dir(const float* __restrict__ Wih, const float* __restrict__ Whh,
                         const float* __restrict__ bih, const float* __restrict__ bhh,
                         const float* __restrict__ Mk,  const float* __restrict__ Mv,
                         __bf16* __restrict__ Wx, __bf16* __restrict__ Wc,
                         __bf16* __restrict__ Wh, float* __restrict__ bias,
                         __bf16* __restrict__ MkP, __bf16* __restrict__ MvP) {
  const int tid = blockIdx.x * blockDim.x + threadIdx.x;
  const int nth = gridDim.x * blockDim.x;
  for (int i = tid; i < G4H * E_DIM; i += nth) {
    const int r = i / E_DIM, c = i % E_DIM;
    Wx[i] = (__bf16)Wih[(size_t)r * (E_DIM + CTX_DIM) + c];
  }
  for (int i = tid; i < G4H * CTXP; i += nth) {
    const int r = i / CTXP, c = i % CTXP;
    Wc[i] = (c < CTX_DIM) ? (__bf16)Wih[(size_t)r * (E_DIM + CTX_DIM) + E_DIM + c]
                          : (__bf16)0.f;
  }
  for (int i = tid; i < G4H * H_DIM; i += nth) Wh[i] = (__bf16)Whh[i];
  for (int i = tid; i < G4H; i += nth) bias[i] = bih[i] + bhh[i];
  for (int i = tid; i < MEMP * H_DIM; i += nth) {
    const int r = i / H_DIM;
    MkP[i] = (r < MEM_DIM) ? (__bf16)Mk[i] : (__bf16)0.f;
  }
  for (int i = tid; i < CTXP * MEMP; i += nth) {
    const int r = i / MEMP, c = i % MEMP;
    MvP[i] = (r < CTX_DIM && c < MEM_DIM) ? (__bf16)Mv[(size_t)r * MEM_DIM + c]
                                          : (__bf16)0.f;
  }
}

__global__ void zero_f32(float* __restrict__ p, int n) {
  const int tid = blockIdx.x * blockDim.x + threadIdx.x;
  const int nth = gridDim.x * blockDim.x;
  for (int i = tid; i < n; i += nth) p[i] = 0.f;
}

// ---------------------------------------------------------------------------
// Per-step: stage x[:, t, :] to bf16, then ctx = softmax(h Mk^T) Mv^T (bf16).
// grid (8 row-blocks, 2 dirs), 256 threads = 8 waves. Always launched;
// s==0 does staging only (ctx buffers are pre-zeroed).
// ---------------------------------------------------------------------------
__global__ void memctx_kernel(const float* __restrict__ xglob,
                              const __bf16* __restrict__ hbfF,
                              const __bf16* __restrict__ hmbfB,
                              const __bf16* __restrict__ MkF, const __bf16* __restrict__ MvF,
                              const __bf16* __restrict__ MkB, const __bf16* __restrict__ MvB,
                              __bf16* __restrict__ ctxF, __bf16* __restrict__ ctxB,
                              __bf16* __restrict__ xbfF, __bf16* __restrict__ xbfB, int s) {
  __shared__ __attribute__((aligned(16))) float  a_f32[16][MEMP];
  __shared__ __attribute__((aligned(16))) __bf16 a_bf[16][MEMP];
  __shared__ __attribute__((aligned(16))) __bf16 h_lds[16][H_DIM];
  __shared__ float red[16][16];
  __shared__ float inv_lds[16];
  const int dir = blockIdx.y;
  const int b0 = blockIdx.x * 16;
  const int wave = threadIdx.x >> 5;
  const int lane = threadIdx.x & 31;
  const int n = lane & 15, half = lane >> 4;

  // ---- stage this WG's 16 batch rows of x at this direction's timestep ----
  {
    const int t = dir ? (T_DIM - 1 - s) : s;
    const float* xsrc = xglob + ((size_t)b0 * T_DIM + t) * E_DIM;
    __bf16* xdst = (dir ? xbfB : xbfF) + (size_t)b0 * E_DIM;
    for (int i = threadIdx.x; i < 16 * (E_DIM / 2); i += 256) {
      const int rowi = i / (E_DIM / 2), col2 = (i % (E_DIM / 2)) * 2;
      const float2 p =
          *reinterpret_cast<const float2*>(xsrc + (size_t)rowi * T_DIM * E_DIM + col2);
      bf16x2 q;
      q.x = (__bf16)p.x;
      q.y = (__bf16)p.y;
      *reinterpret_cast<bf16x2*>(xdst + (size_t)rowi * E_DIM + col2) = q;
    }
  }
  if (s == 0) return;  // uniform: step 0 uses the zero dummy context

  const __bf16* hsrc = dir ? hmbfB : (hbfF + (size_t)(s & 1) * B_DIM * H_DIM);
  const __bf16* Mk = dir ? MkB : MkF;
  const __bf16* Mv = dir ? MvB : MvF;
  __bf16* ctx = dir ? ctxB : ctxF;

  // stage the 16 h rows once (shared across all 32 N-tiles of phase 1)
  for (int i = threadIdx.x; i < 16 * (H_DIM / 8); i += 256) {
    const int r = i / (H_DIM / 8), c8 = (i % (H_DIM / 8)) * 8;
    *reinterpret_cast<bf16x8*>(&h_lds[r][c8]) =
        *reinterpret_cast<const bf16x8*>(hsrc + (size_t)(b0 + r) * H_DIM + c8);
  }
  __syncthreads();

  // phase 1: a = exp(h Mk^T); pad columns (>=MEM) masked to 0
#pragma unroll
  for (int q = 0; q < 4; ++q) {
    const int nt = wave + 8 * q;
    v8f acc = {};
#pragma unroll 4
    for (int kk = 0; kk < H_DIM / 32; ++kk) {
      const v16bf A  = load_frag_bf16(&h_lds[0][0] + kk * 32, H_DIM);
      const v16bf Bm = load_frag_bf16(Mk + (size_t)(nt * 16) * H_DIM + kk * 32, H_DIM);
      acc = wmma_bf16(A, Bm, acc);
    }
    const int col = nt * 16 + n;
#pragma unroll
    for (int r = 0; r < 8; ++r) {
      const int m = r + 8 * half;
      const float e = (col < MEM_DIM) ? __expf(acc[r]) : 0.f;
      a_f32[m][col] = e;
      a_bf[m][col] = (__bf16)e;
    }
  }
  __syncthreads();

  // phase 2: per-row reciprocal sums (fp32)
  {
    const int row = threadIdx.x >> 4, part = threadIdx.x & 15;
    float sum = 0.f;
#pragma unroll 8
    for (int c = part * 32; c < part * 32 + 32; ++c) sum += a_f32[row][c];
    red[row][part] = sum;
  }
  __syncthreads();
  if (threadIdx.x < 16) {
    float s2 = 0.f;
#pragma unroll
    for (int p = 0; p < 16; ++p) s2 += red[threadIdx.x][p];
    inv_lds[threadIdx.x] = 1.f / s2;
  }
  __syncthreads();

  // phase 3: ctx = (a * inv) @ Mv^T, normalization folded into bf16 store
  for (int q = 0; q < 3; ++q) {
    const int nt = wave + 8 * q;
    if (nt >= CTXP / 16) break;  // wave-uniform
    v8f acc = {};
#pragma unroll 4
    for (int kk = 0; kk < MEMP / 32; ++kk) {
      const v16bf A  = load_frag_bf16(&a_bf[0][0] + kk * 32, MEMP);
      const v16bf Bm = load_frag_bf16(Mv + (size_t)(nt * 16) * MEMP + kk * 32, MEMP);
      acc = wmma_bf16(A, Bm, acc);
    }
    const int col = nt * 16 + n;
#pragma unroll
    for (int r = 0; r < 8; ++r) {
      const int m = r + 8 * half;
      ctx[(size_t)(b0 + m) * CTXP + col] = (__bf16)(acc[r] * inv_lds[m]);
    }
  }
}

// ---------------------------------------------------------------------------
// One LSTM step, both directions: g = x Wx^T + h Wh^T + ctx Wc^T + b, cell.
// grid (8 b-tiles, 16 j-tile-pairs, 2 dirs), 256 threads = 8 waves:
// wave = {jhalf, gate}; activation rows staged once in LDS for all 8 waves.
// bf16 hidden mirror ping-pongs on step parity to avoid intra-launch races.
// ---------------------------------------------------------------------------
__global__ void lstm_step_kernel(const int* __restrict__ lens,
    const __bf16* __restrict__ WxF, const __bf16* __restrict__ WcF,
    const __bf16* __restrict__ WhF, const float* __restrict__ biasF,
    const __bf16* __restrict__ WxB, const __bf16* __restrict__ WcB,
    const __bf16* __restrict__ WhB, const float* __restrict__ biasB,
    const __bf16* __restrict__ xbfF, const __bf16* __restrict__ xbfB,
    __bf16* __restrict__ hbfF, float* __restrict__ cF, const __bf16* __restrict__ ctxF,
    __bf16* __restrict__ hbfB, float* __restrict__ cB, const __bf16* __restrict__ ctxB,
    float* __restrict__ hmB, __bf16* __restrict__ hmbfB,
    float* __restrict__ out, int s) {
  __shared__ __attribute__((aligned(16))) __bf16 sA[16][AK];  // x | h | ctx rows
  __shared__ float g_lds[2][4][16][16];
  const int b0 = blockIdx.x * 16, j0 = blockIdx.y * 32, dir = blockIdx.z;
  const int wave = threadIdx.x >> 5, lane = threadIdx.x & 31;
  const int gate = wave & 3, jhalf = wave >> 2;
  const int n = lane & 15, half = lane >> 4;
  const int t = dir ? (T_DIM - 1 - s) : s;

  const __bf16* Wx = dir ? WxB : WxF;
  const __bf16* Wc = dir ? WcB : WcF;
  const __bf16* Wh = dir ? WhB : WhF;
  const float* bias = dir ? biasB : biasF;
  const __bf16* ctx = dir ? ctxB : ctxF;
  const __bf16* xbf = dir ? xbfB : xbfF;
  const __bf16* hprev = (dir ? hbfB : hbfF) + (size_t)(s & 1) * B_DIM * H_DIM;
  __bf16* hnext       = (dir ? hbfB : hbfF) + (size_t)((s & 1) ^ 1) * B_DIM * H_DIM;
  float* cbuf = dir ? cB : cF;

  // ---- stage the 16 activation rows (x, h, ctx) once for all 8 waves ----
  for (int i = threadIdx.x; i < 16 * (E_DIM / 8); i += 256) {
    const int r = i / (E_DIM / 8), c8 = (i % (E_DIM / 8)) * 8;
    *reinterpret_cast<bf16x8*>(&sA[r][c8]) =
        *reinterpret_cast<const bf16x8*>(xbf + (size_t)(b0 + r) * E_DIM + c8);
  }
  for (int i = threadIdx.x; i < 16 * (H_DIM / 8); i += 256) {
    const int r = i / (H_DIM / 8), c8 = (i % (H_DIM / 8)) * 8;
    *reinterpret_cast<bf16x8*>(&sA[r][E_DIM + c8]) =
        *reinterpret_cast<const bf16x8*>(hprev + (size_t)(b0 + r) * H_DIM + c8);
  }
  for (int i = threadIdx.x; i < 16 * (CTXP / 8); i += 256) {
    const int r = i / (CTXP / 8), c8 = (i % (CTXP / 8)) * 8;
    *reinterpret_cast<bf16x8*>(&sA[r][E_DIM + H_DIM + c8]) =
        *reinterpret_cast<const bf16x8*>(ctx + (size_t)(b0 + r) * CTXP + c8);
  }
  __syncthreads();

  const int nw = gate * H_DIM + j0 + jhalf * 16;  // weight-row base for this wave
  __builtin_prefetch(Wx + (size_t)nw * E_DIM, 0, 0);
  __builtin_prefetch(Wh + (size_t)nw * H_DIM, 0, 0);
  __builtin_prefetch(Wc + (size_t)nw * CTXP, 0, 0);

  v8f acc = {};
  // x part (K = E)
#pragma unroll 4
  for (int kk = 0; kk < E_DIM / 32; ++kk) {
    const v16bf A  = load_frag_bf16(&sA[0][0] + kk * 32, AK);
    const v16bf Bw = load_frag_bf16(Wx + (size_t)nw * E_DIM + kk * 32, E_DIM);
    acc = wmma_bf16(A, Bw, acc);
  }
  // h part (K = H)
#pragma unroll 4
  for (int kk = 0; kk < H_DIM / 32; ++kk) {
    const v16bf A  = load_frag_bf16(&sA[0][E_DIM] + kk * 32, AK);
    const v16bf Bw = load_frag_bf16(Wh + (size_t)nw * H_DIM + kk * 32, H_DIM);
    acc = wmma_bf16(A, Bw, acc);
  }
  // ctx part (K = CTXP; pad region zero on both operands)
#pragma unroll 4
  for (int kk = 0; kk < CTXP / 32; ++kk) {
    const v16bf A  = load_frag_bf16(&sA[0][E_DIM + H_DIM] + kk * 32, AK);
    const v16bf Bw = load_frag_bf16(Wc + (size_t)nw * CTXP + kk * 32, CTXP);
    acc = wmma_bf16(A, Bw, acc);
  }

  const float bval = bias[nw + n];
#pragma unroll
  for (int r = 0; r < 8; ++r) g_lds[jhalf][gate][r + 8 * half][n] = acc[r] + bval;
  __syncthreads();

  // fused LSTM cell over the 16x32 tile (2 elements / thread)
#pragma unroll
  for (int e = threadIdx.x; e < 512; e += 256) {
    const int m = e >> 5, jl = e & 31;
    const int js = jl >> 4, nn = jl & 15;
    const float gi = g_lds[js][0][m][nn], gf = g_lds[js][1][m][nn];
    const float gg = g_lds[js][2][m][nn], go = g_lds[js][3][m][nn];
    const int b = b0 + m, j = j0 + jl;
    const size_t idx = (size_t)b * H_DIM + j;
    const float cold = cbuf[idx];
    const float cnew = sigm(gf) * cold + sigm(gi) * tanhf(gg);
    const float hn = sigm(go) * tanhf(cnew);
    cbuf[idx] = cnew;
    hnext[idx] = (__bf16)hn;
    const int len = lens[b];
    if (dir == 0) {
      if (len - 1 == t) out[(size_t)b * (2 * H_DIM) + 2 * j] = hn;  // final fwd hidden
    } else {
      const float hm = (t < len) ? hn : 0.f;  // masked hidden feeds next memory read
      hmB[idx] = hm;
      hmbfB[idx] = (__bf16)hm;
    }
  }
}

__global__ void finalize_kernel(const float* __restrict__ hmB, float* __restrict__ out) {
  const int i = blockIdx.x * blockDim.x + threadIdx.x;
  if (i < B_DIM * H_DIM) {
    const int b = i / H_DIM, j = i % H_DIM;
    out[(size_t)b * (2 * H_DIM) + 2 * j + 1] = hmB[i];
  }
}

// ---------------------------------------------------------------------------
extern "C" void kernel_launch(void* const* d_in, const int* in_sizes, int n_in,
                              void* d_out, int out_size, void* d_ws, size_t ws_size,
                              hipStream_t stream) {
  (void)in_sizes; (void)n_in; (void)out_size; (void)ws_size;
  const float* x      = (const float*)d_in[0];
  const int*   lens   = (const int*)d_in[1];
  const float* Wih_f  = (const float*)d_in[2];
  const float* Whh_f  = (const float*)d_in[3];
  const float* bih_f  = (const float*)d_in[4];
  const float* bhh_f  = (const float*)d_in[5];
  const float* Wih_b  = (const float*)d_in[6];
  const float* Whh_b  = (const float*)d_in[7];
  const float* bih_b  = (const float*)d_in[8];
  const float* bhh_b  = (const float*)d_in[9];
  const float* Mk_f   = (const float*)d_in[10];
  const float* Mv_f   = (const float*)d_in[11];
  const float* Mk_b   = (const float*)d_in[12];
  const float* Mv_b   = (const float*)d_in[13];
  float* out = (float*)d_out;

  char* ws = (char*)d_ws;
  size_t off = 0;
  auto alloc = [&](size_t bytes) -> void* {
    void* p = ws + off;
    off = (off + bytes + 255) & ~(size_t)255;
    return p;
  };
  const size_t BH = (size_t)B_DIM * H_DIM;

  __bf16* WxF  = (__bf16*)alloc((size_t)G4H * E_DIM * 2);
  __bf16* WcF  = (__bf16*)alloc((size_t)G4H * CTXP * 2);
  __bf16* WhF  = (__bf16*)alloc((size_t)G4H * H_DIM * 2);
  float*  bF   = (float*)alloc((size_t)G4H * 4);
  __bf16* MkPF = (__bf16*)alloc((size_t)MEMP * H_DIM * 2);
  __bf16* MvPF = (__bf16*)alloc((size_t)CTXP * MEMP * 2);
  __bf16* WxB  = (__bf16*)alloc((size_t)G4H * E_DIM * 2);
  __bf16* WcB  = (__bf16*)alloc((size_t)G4H * CTXP * 2);
  __bf16* WhB  = (__bf16*)alloc((size_t)G4H * H_DIM * 2);
  float*  bB   = (float*)alloc((size_t)G4H * 4);
  __bf16* MkPB = (__bf16*)alloc((size_t)MEMP * H_DIM * 2);
  __bf16* MvPB = (__bf16*)alloc((size_t)CTXP * MEMP * 2);

  char* state_begin = ws + off;
  __bf16* hbfF = (__bf16*)alloc(2 * BH * 2);
  float*  cF   = (float*)alloc(BH * 4);
  __bf16* hbfB = (__bf16*)alloc(2 * BH * 2);
  float*  cB   = (float*)alloc(BH * 4);
  float*  hmB  = (float*)alloc(BH * 4);
  __bf16* hmbfB = (__bf16*)alloc(BH * 2);
  __bf16* ctxF = (__bf16*)alloc((size_t)B_DIM * CTXP * 2);
  __bf16* ctxB = (__bf16*)alloc((size_t)B_DIM * CTXP * 2);
  __bf16* xbfF = (__bf16*)alloc((size_t)B_DIM * E_DIM * 2);
  __bf16* xbfB = (__bf16*)alloc((size_t)B_DIM * E_DIM * 2);
  const int state_n = (int)(((ws + off) - state_begin) / 4);

  pack_dir<<<dim3(512), 256, 0, stream>>>(Wih_f, Whh_f, bih_f, bhh_f, Mk_f, Mv_f,
                                          WxF, WcF, WhF, bF, MkPF, MvPF);
  pack_dir<<<dim3(512), 256, 0, stream>>>(Wih_b, Whh_b, bih_b, bhh_b, Mk_b, Mv_b,
                                          WxB, WcB, WhB, bB, MkPB, MvPB);
  zero_f32<<<dim3((state_n + 255) / 256), 256, 0, stream>>>((float*)state_begin, state_n);

  for (int s = 0; s < T_DIM; ++s) {
    memctx_kernel<<<dim3(8, 2), 256, 0, stream>>>(x, hbfF, hmbfB, MkPF, MvPF, MkPB,
                                                  MvPB, ctxF, ctxB, xbfF, xbfB, s);
    lstm_step_kernel<<<dim3(8, 16, 2), 256, 0, stream>>>(
        lens, WxF, WcF, WhF, bF, WxB, WcB, WhB, bB, xbfF, xbfB,
        hbfF, cF, ctxF, hbfB, cB, ctxB, hmB, hmbfB, out, s);
  }
  finalize_kernel<<<dim3((int)((BH + 255) / 256)), 256, 0, stream>>>(hmB, out);
}
